// GraphAttentionLayer2D_61435212202108
// MI455X (gfx1250) — matrix operand order repaired and checked by reading, not compiled, not run on hardware
//
#include <hip/hip_runtime.h>
#include <hip/hip_bf16.h>
#include <cstdint>

// ---------------------------------------------------------------------------
// GraphAttentionLayer2D on MI455X (gfx1250):
//   3 bf16 WMMA GEMM stages (f32 accum) fed by the Tensor Data Mover
//   (tensor_load_to_lds -> LDS double buffer -> ds_load fragments -> v_wmma),
//   fused leaky-relu+softmax(C=32), adj normalization, ELU epilogue.
// ---------------------------------------------------------------------------

typedef __attribute__((ext_vector_type(16))) __bf16 v16bf;
typedef __attribute__((ext_vector_type(8)))  __bf16 v8bf;
typedef __attribute__((ext_vector_type(8)))  float  v8f;
typedef __attribute__((ext_vector_type(4)))  unsigned v4u;
typedef __attribute__((ext_vector_type(8)))  int      v8i;
typedef __attribute__((ext_vector_type(4)))  int      v4i;

constexpr int Nn = 8, Cc = 32, Tt = 256, Vv = 256, Ff = 256;
constexpr int NC = Nn * Cc;               // 256 batches
constexpr float LRELU_ALPHA = 0.2f;

#define TILE_M 64
#define TILE_N 256
#define TILE_K 32

#if __has_builtin(__builtin_amdgcn_tensor_load_to_lds) && \
    __has_builtin(__builtin_amdgcn_s_wait_tensorcnt)
#define USE_TDM 1
#else
#define USE_TDM 0
#endif

__device__ inline __bf16 f2bf(float x) {
  union { float f; unsigned u; } a; a.f = x;
  unsigned r = a.u + 0x7FFFu + ((a.u >> 16) & 1u);   // round-to-nearest-even
  union { unsigned short s; __bf16 b; } o; o.s = (unsigned short)(r >> 16);
  return o.b;
}
__device__ inline float bf2f(__bf16 x) {
  union { unsigned short s; __bf16 b; } i; i.b = x;
  union { unsigned u; float f; } o; o.u = ((unsigned)i.s) << 16;
  return o.f;
}

// --- Tensor Data Mover: 2-D tile (tile_h rows x 32 bf16) global -> LDS -----
// D# bitfields per CDNA5 ISA.
// This toolchain exposes the 6-arg builtin: (g0, g1, g2, g3, g_extra, cpol).
#if USE_TDM
__device__ inline void tdm_load_2d(const __bf16* gptr, unsigned lds_byte_addr,
                                   unsigned tensor_w, unsigned rows,
                                   unsigned tile_w, unsigned tile_h) {
  unsigned long long ga = (unsigned long long)(uintptr_t)gptr;
  v4u g0;
  g0[0] = 1u;                                          // count=1, user mode
  g0[1] = lds_byte_addr;                               // lds_addr [63:32]
  g0[2] = (unsigned)(ga & 0xFFFFFFFFull);              // global_addr lo
  g0[3] = (unsigned)((ga >> 32) & 0x01FFFFFFull)       // global_addr hi
        | (2u << 30);                                  // type=2 ("image")
  v8i g1;
  g1[0] = (int)(1u << 16);                             // wg_mask=0, data_size=2B
  g1[1] = (int)((tensor_w & 0xFFFFu) << 16);           // tensor_dim0 lo16
  g1[2] = (int)((tensor_w >> 16) | ((rows & 0xFFFFu) << 16)); // dim0 hi | dim1 lo
  g1[3] = (int)((rows >> 16) | ((tile_w & 0xFFFFu) << 16));   // dim1 hi | tile_dim0
  g1[4] = (int)(tile_h & 0xFFFFu);                     // tile_dim1 | tile_dim2=0
  g1[5] = (int)tensor_w;                               // tensor_dim0_stride lo32
  g1[6] = 0;                                           // stride0 hi | stride1 lo
  g1[7] = 0;                                           // stride1 hi
  v4i gz  = {0, 0, 0, 0};                              // groups 2/3: 2-D tensor
  v8i gzz = {0, 0, 0, 0, 0, 0, 0, 0};                  // extra group (unused)
  __builtin_amdgcn_tensor_load_to_lds(g0, g1, gz, gz, gzz, 0);
}
#endif

// --- WMMA fragment loads from LDS tiles (CDNA5 ISA 7.12.2 layouts) ----------
__device__ inline v16bf lds_a_frag(const __bf16 (&Ab)[TILE_M][TILE_K], int m_local, int lane) {
  int m  = m_local + (lane & 15);
  int kb = (lane >> 4) ? 8 : 0;   // lanes<16: K {0..7,16..23}; lanes>=16: +8
  union { v16bf v; v8bf h[2]; } f;
  f.h[0] = *(const v8bf*)&Ab[m][kb];
  f.h[1] = *(const v8bf*)&Ab[m][kb + 16];
  return f.v;
}
__device__ inline v16bf lds_b_frag(const __bf16 (&Bb)[TILE_N][TILE_K], int n_local, int lane) {
  int n  = n_local + (lane & 15);
  int kb = (lane >> 4) ? 8 : 0;   // lanes<16: K {0..7,16..23}; lanes>=16: +8
  union { v16bf v; v8bf h[2]; } f;
  f.h[0] = *(const v8bf*)&Bb[n][kb];
  f.h[1] = *(const v8bf*)&Bb[n][kb + 16];
  return f.v;
}

// --- transpose f32 -> bf16, dst[cols][rows] --------------------------------
__global__ void __launch_bounds__(256)
transpose_f32_to_bf16(const float* __restrict__ src, __bf16* __restrict__ dst,
                      int rows, int cols, size_t batchStride) {
  __shared__ float tile[32][33];
  int batch = blockIdx.z;
  const float* S = src + (size_t)batch * batchStride;
  __bf16* D = dst + (size_t)batch * batchStride;
  int c0 = blockIdx.x * 32;
  int r0 = blockIdx.y * 32;
  int tx = threadIdx.x, ty = threadIdx.y;       // (32, 8)
  #pragma unroll
  for (int i = 0; i < 4; i++) {
    int r = ty + i * 8;
    tile[r][tx] = S[(size_t)(r0 + r) * cols + c0 + tx];
  }
  __syncthreads();
  #pragma unroll
  for (int i = 0; i < 4; i++) {
    int r = ty + i * 8;
    D[(size_t)(c0 + r) * rows + r0 + tx] = f2bf(tile[tx][r]);
  }
}

// --- adjacency normalization: adjT[v][i] = bf16(adj_norm[i][v]) ------------
__global__ void __launch_bounds__(256)
adj_kernel(const float* __restrict__ B_adj, __bf16* __restrict__ adjT) {
  __shared__ float s0[256], s1[256], sd[256];
  int t = threadIdx.x;
  float mn = 1e30f, mx = -1e30f;
  for (int idx = t; idx < Vv * Vv; idx += 256) {
    int r = idx >> 8, c = idx & 255;
    float v = B_adj[idx] + (r == c ? 1.0f : 0.0f);
    mn = fminf(mn, v); mx = fmaxf(mx, v);
  }
  s0[t] = mn; s1[t] = mx; __syncthreads();
  for (int s = 128; s > 0; s >>= 1) {
    if (t < s) { s0[t] = fminf(s0[t], s0[t + s]); s1[t] = fmaxf(s1[t], s1[t + s]); }
    __syncthreads();
  }
  float gmn = s0[0];
  float inv = 1.0f / (s1[0] - gmn);
  float rs = 0.f;
  for (int c = 0; c < Vv; c++) {
    float v = B_adj[t * Vv + c] + (t == c ? 1.0f : 0.0f);
    rs += (v - gmn) * inv;
  }
  sd[t] = rsqrtf(rs);
  __syncthreads();
  for (int c = 0; c < Vv; c++) {
    float v = B_adj[t * Vv + c] + (t == c ? 1.0f : 0.0f);
    float an = (v - gmn) * inv * sd[t] * sd[c];
    adjT[(size_t)c * Vv + t] = f2bf(an);      // i = t (row), v = c (col)
  }
}

// --- batched GEMM: TDM-staged LDS double buffer + WMMA ----------------------
// D = A(MxK row-major) x B (given transposed: NxK row-major), N == 256.
// mode 0: store D^T as bf16 (Dbf[n][m], ld=M)  -> next stage's B^T
// mode 1: store ELU(D) as f32 (Dout[m][n], ld=N)
__global__ void __launch_bounds__(256)
gemm_bf16_wmma(const __bf16* __restrict__ Aall, const __bf16* __restrict__ Ball,
               __bf16* __restrict__ Dbf, float* __restrict__ Dout,
               int M, int N, int K,
               size_t strideA, size_t strideB, size_t strideD, int mode) {
  __shared__ __bf16 Ab[2][TILE_M][TILE_K];   //  8 KB
  __shared__ __bf16 Bb[2][TILE_N][TILE_K];   // 32 KB

  int batch = blockIdx.y;
  const __bf16* Atile = Aall + (size_t)batch * strideA + (size_t)blockIdx.x * TILE_M * K;
  const __bf16* Bp    = Ball + (size_t)batch * strideB;

  int lane = threadIdx.x & 31;
  int wid  = threadIdx.x >> 5;
  int mw = wid & 1, nw = wid >> 1;               // 2 waves x M, 4 waves x N

  v8f zero = {};
  v8f acc[2][4];
  #pragma unroll
  for (int i = 0; i < 2; i++)
    #pragma unroll
    for (int j = 0; j < 4; j++) acc[i][j] = zero;

  // ---- stage a k-slice into LDS buffer nb ----
  auto stage = [&](int nb, int kk) {
#if USE_TDM
    if (wid == 0) {
      tdm_load_2d(Atile + kk, (unsigned)(uintptr_t)&Ab[nb][0][0],
                  (unsigned)K, TILE_M, TILE_K, TILE_M);
      tdm_load_2d(Bp + kk, (unsigned)(uintptr_t)&Bb[nb][0][0],
                  (unsigned)K, TILE_N, TILE_K, TILE_N);
    }
#else
    int t = threadIdx.x;
    {   // A tile: 64 rows x 32 = 256 16B-chunks, one per thread
      int rowA = t >> 2, ca = (t & 3) * 8;
      *(v8bf*)&Ab[nb][rowA][ca] = *(const v8bf*)(Atile + (size_t)rowA * K + kk + ca);
    }
    #pragma unroll
    for (int r = 0; r < 4; r++) {  // B tile: 256 rows x 32 = 1024 chunks
      int ch = t + 256 * r;
      int rowB = ch >> 2, cb = (ch & 3) * 8;
      *(v8bf*)&Bb[nb][rowB][cb] = *(const v8bf*)(Bp + (size_t)rowB * K + kk + cb);
    }
#endif
  };
  auto wait_fill = [&]() {
#if USE_TDM
    if (wid == 0) __builtin_amdgcn_s_wait_tensorcnt(0);
#endif
    __syncthreads();
  };

  stage(0, 0);
  wait_fill();

  int nb = 0;
  for (int kk = 0; kk < K; kk += TILE_K) {
    if (kk + TILE_K < K) stage(nb ^ 1, kk + TILE_K);   // overlap DMA with WMMA

    v16bf af[2], bfg[4];
    #pragma unroll
    for (int mt = 0; mt < 2; mt++) af[mt] = lds_a_frag(Ab[nb], mw * 32 + mt * 16, lane);
    #pragma unroll
    for (int nt = 0; nt < 4; nt++) bfg[nt] = lds_b_frag(Bb[nb], nw * 64 + nt * 16, lane);
    #pragma unroll
    for (int mt = 0; mt < 2; mt++)
      #pragma unroll
      for (int nt = 0; nt < 4; nt++)
        acc[mt][nt] = __builtin_amdgcn_wmma_f32_16x16x32_bf16(
            false, af[mt], false, bfg[nt], (short)0, acc[mt][nt], false, false);

    wait_fill();
    nb ^= 1;
  }

  // ---- epilogue ----
  int mb   = (lane >> 4) ? 8 : 0;
  int ncol = lane & 15;
  int m_blk = blockIdx.x * TILE_M + mw * 32;
  if (mode == 0) {
    __bf16* D = Dbf + (size_t)batch * strideD;
    #pragma unroll
    for (int mt = 0; mt < 2; mt++)
      #pragma unroll
      for (int nt = 0; nt < 4; nt++) {
        int n  = nw * 64 + nt * 16 + ncol;
        int m0 = m_blk + mt * 16 + mb;
        v8bf o;
        #pragma unroll
        for (int r = 0; r < 8; r++) o[r] = f2bf(acc[mt][nt][r]);
        *(v8bf*)(D + (size_t)n * M + m0) = o;     // 16B contiguous store
      }
  } else {
    float* O = Dout + (size_t)batch * strideD;
    #pragma unroll
    for (int mt = 0; mt < 2; mt++)
      #pragma unroll
      for (int nt = 0; nt < 4; nt++) {
        int n  = nw * 64 + nt * 16 + ncol;
        int m0 = m_blk + mt * 16 + mb;
        #pragma unroll
        for (int r = 0; r < 8; r++) {
          float x = acc[mt][nt][r];
          x = x > 0.f ? x : expm1f(x);            // ELU (alpha = 1)
          O[(size_t)(m0 + r) * N + n] = x;
        }
      }
  }
}

// --- f1/f2: f{1,2}[(n*C+c)*V + v] = sum_f WhT[b][f][v] * a{1,2}[f] ---------
__global__ void __launch_bounds__(256)
f1f2_kernel(const __bf16* __restrict__ WhT, const float* __restrict__ a,
            float* __restrict__ f1, float* __restrict__ f2) {
  int batch = blockIdx.x;
  int v = threadIdx.x;
  const __bf16* p = WhT + (size_t)batch * Ff * Vv + v;
  float s1 = 0.f, s2 = 0.f;
  for (int f = 0; f < Ff; f++) {
    float w = bf2f(p[(size_t)f * Vv]);
    s1 += w * a[f];
    s2 += w * a[Ff + f];
  }
  f1[(size_t)batch * Vv + v] = s1;
  f2[(size_t)batch * Vv + v] = s2;
}

// --- att[b][i][j] (bf16): softmax over c of leaky(f1[b,i]+f2[b,j]) ---------
__global__ void __launch_bounds__(256)
att_kernel(const float* __restrict__ f1, const float* __restrict__ f2,
           __bf16* __restrict__ att) {
  int j = threadIdx.x;
  int i = blockIdx.x;
  int n = blockIdx.y;
  float ec[Cc];
  float m = -1e30f;
  #pragma unroll
  for (int c = 0; c < Cc; c++) {
    size_t b = (size_t)(n * Cc + c);
    float e = f1[b * Vv + i] + f2[b * Vv + j];
    e = e >= 0.f ? e : LRELU_ALPHA * e;
    ec[c] = e;
    m = fmaxf(m, e);
  }
  float s = 0.f;
  #pragma unroll
  for (int c = 0; c < Cc; c++) { ec[c] = __expf(ec[c] - m); s += ec[c]; }
  float inv = 1.0f / s;
  #pragma unroll
  for (int c = 0; c < Cc; c++) {
    size_t b = (size_t)(n * Cc + c);
    att[(b * Vv + i) * Vv + j] = f2bf(ec[c] * inv);
  }
}

// ---------------------------------------------------------------------------
extern "C" void kernel_launch(void* const* d_in, const int* in_sizes, int n_in,
                              void* d_out, int out_size, void* d_ws, size_t ws_size,
                              hipStream_t stream) {
  const float* h     = (const float*)d_in[0];   // (N,C,T,V) f32
  const float* W     = (const float*)d_in[1];   // (T,F) f32
  const float* a     = (const float*)d_in[2];   // (2F,1) f32
  const float* B_adj = (const float*)d_in[3];   // (V,V) f32
  float* out = (float*)d_out;                   // (N,C,F,V) f32

  // workspace carve-up (~135 MB)
  char* ws = (char*)d_ws;
  size_t off = 0;
  const size_t big = (size_t)NC * Vv * Tt * sizeof(__bf16);   // 33.5 MB each
  __bf16* hT    = (__bf16*)(ws + off); off += big;            // [b][v][t]
  __bf16* WhT   = (__bf16*)(ws + off); off += big;            // [b][f][v]
  __bf16* att   = (__bf16*)(ws + off); off += big;            // [b][i][j]
  __bf16* out1T = (__bf16*)(ws + off); off += big;            // [b][e][i]
  __bf16* WT    = (__bf16*)(ws + off); off += (size_t)Ff * Tt * sizeof(__bf16);
  __bf16* adjT  = (__bf16*)(ws + off); off += (size_t)Vv * Vv * sizeof(__bf16);
  float*  f1    = (float*)(ws + off);  off += (size_t)NC * Vv * sizeof(float);
  float*  f2    = (float*)(ws + off);  off += (size_t)NC * Vv * sizeof(float);

  dim3 tb(32, 8);
  // hT[b][v][t] = h[b][t][v]   (b = n*C+c)
  transpose_f32_to_bf16<<<dim3(Vv / 32, Tt / 32, NC), tb, 0, stream>>>(
      h, hT, Tt, Vv, (size_t)Tt * Vv);
  // WT[f][t] = W[t][f]
  transpose_f32_to_bf16<<<dim3(Ff / 32, Tt / 32, 1), tb, 0, stream>>>(
      W, WT, Tt, Ff, (size_t)Tt * Ff);
  adj_kernel<<<1, 256, 0, stream>>>(B_adj, adjT);

  // Stage 1: WhT[b][f][v] = (hT[b] (VxT) x W (TxF))^T
  gemm_bf16_wmma<<<dim3(Vv / TILE_M, NC), 256, 0, stream>>>(
      hT, WT, WhT, nullptr, Vv, Ff, Tt,
      (size_t)Vv * Tt, 0, (size_t)Ff * Vv, 0);

  f1f2_kernel<<<NC, 256, 0, stream>>>(WhT, a, f1, f2);
  att_kernel<<<dim3(Vv, Nn), 256, 0, stream>>>(f1, f2, att);

  // Stage 2: out1T[b][e][i] = (att[b] (IxJ) x Wh[b] (JxE))^T
  gemm_bf16_wmma<<<dim3(Vv / TILE_M, NC), 256, 0, stream>>>(
      att, WhT, out1T, nullptr, Vv, Ff, Vv,
      (size_t)Vv * Vv, (size_t)Ff * Vv, (size_t)Ff * Vv, 0);

  // Stage 3: out[b][e][v] = elu( out1T[b] (ExI) x adj_norm (IxV) )
  gemm_bf16_wmma<<<dim3(Ff / TILE_M, NC), 256, 0, stream>>>(
      out1T, adjT, nullptr, out, Ff, Vv, Vv,
      (size_t)Ff * Vv, 0, (size_t)Ff * Vv, 1);
}
